// VariableLSTMModel_79980880986877
// MI455X (gfx1250) — matrix operand order repaired and checked by reading, not compiled, not run on hardware
//
#include <hip/hip_runtime.h>

// Problem constants
#define B_   64
#define T_   4096
#define H_   128
#define G4H_ 512
#define INC_ 64
#define L1_  256
#define L2_  256

typedef _Float16 v16h __attribute__((ext_vector_type(16)));
typedef _Float16 v8h  __attribute__((ext_vector_type(8)));
typedef float    v8f  __attribute__((ext_vector_type(8)));

union F8 { v8f v; float f[8]; };

__device__ __forceinline__ float sigf(float x) { return 1.0f / (1.0f + __expf(-x)); }
__device__ __forceinline__ float eluf(float x) { return x > 0.0f ? x : (__expf(x) - 1.0f); }

// A-operand tile (16x32 f16, M = lane%16): elements 0..7 <- K = kb + lh*8 + 0..7,
// elements 8..15 <- K = kb + 16 + lh*8 + 0..7  (per ISA 16-bit A layout table).
__device__ __forceinline__ v16h load_a_tile(const _Float16* p) {
  v8h lo = *(const v8h*)(p);
  v8h hi = *(const v8h*)(p + 16);
  return __builtin_shufflevector(lo, hi, 0,1,2,3,4,5,6,7,8,9,10,11,12,13,14,15);
}

// CDNA5 async global->LDS copy: 32 bytes per lane (two b128 transfers).
// GVS mode: mem = SGPR64 + VGPR32 + IOFFSET; IOFFSET also offsets the LDS dest.
__device__ __forceinline__ void async_copy32(unsigned lds_addr, unsigned voff,
                                             unsigned long long gbase) {
  asm volatile(
      "global_load_async_to_lds_b128 %0, %1, %2\n\t"
      "global_load_async_to_lds_b128 %0, %1, %2 offset:16"
      :: "v"(lds_addr), "v"(voff), "s"(gbase) : "memory");
}
__device__ __forceinline__ void async_wait0() {
  asm volatile("s_wait_asynccnt 0x0" ::: "memory");
}

// ---------------------------------------------------------------------------
// Kernel 1: per-batch mean / std over T, plus transposed normalized input
// xnT[t][b] so the recurrence kernel reads 64 contiguous floats per step.
// ---------------------------------------------------------------------------
__global__ void stats_kernel(const float* __restrict__ x,
                             float* __restrict__ mu, float* __restrict__ sd,
                             float* __restrict__ rstd, float* __restrict__ xnT) {
  __shared__ float r1[256], r2[256];
  __shared__ float smu, srs;
  const int b = blockIdx.x, tid = threadIdx.x;
  float s1 = 0.f, s2 = 0.f;
  for (int t = tid; t < T_; t += 256) {
    float v = x[b * T_ + t];
    s1 += v; s2 += v * v;
  }
  r1[tid] = s1; r2[tid] = s2;
  __syncthreads();
  for (int off = 128; off > 0; off >>= 1) {
    if (tid < off) { r1[tid] += r1[tid + off]; r2[tid] += r2[tid + off]; }
    __syncthreads();
  }
  if (tid == 0) {
    float m   = r1[0] * (1.0f / T_);
    float var = fmaxf(r2[0] * (1.0f / T_) - m * m, 0.0f);
    float sv  = sqrtf(var);
    float rs  = 1.0f / sv;
    mu[b] = m; sd[b] = sv; rstd[b] = rs;
    smu = m; srs = rs;
  }
  __syncthreads();
  const float m = smu, rs = srs;
  for (int t = tid; t < T_; t += 256)
    xnT[t * B_ + b] = (x[b * T_ + t] - m) * rs;
}

// ---------------------------------------------------------------------------
// Kernel 2: one persistent workgroup runs a full LSTM layer.
//   LAYER==0: gates = xn[b,t]*Wx0 + h@Wh0 + b0           (K = 128, rank-1 input)
//   LAYER==1: gates = [h1_prev | h0_t] @ [Wh1;Wx1] + b1  (K = 256, input fused)
// 16 waves: wave w -> j-slice s = w>>1 (16 hidden cols, same slice in all 4
// gate chunks) and batch half mh = w&1 (2 M-tiles of 16). Gate combine is
// in-register; h is exchanged via LDS with 2 barriers/step; c stays in VGPRs.
// Layer 1 streams h0_t into LDS with GLOBAL_LOAD_ASYNC_TO_LDS (ASYNCcnt).
// ---------------------------------------------------------------------------
template <int LAYER>
__global__ void lstm_kernel(const float* __restrict__ xnT,
                            const _Float16* __restrict__ h0in,
                            const float* __restrict__ Wx,
                            const float* __restrict__ Wh,
                            const float* __restrict__ bias,
                            _Float16* __restrict__ h0out,
                            float* __restrict__ feat) {
  constexpr int KR = (LAYER == 0) ? 128 : 256;   // K of recurrent GEMM
  constexpr int KT = KR / 32;                    // k-tiles of 32
  extern __shared__ _Float16 smem[];
  _Float16* whT  = smem;                         // [512][KR] f16, K-contiguous
  _Float16* hbuf = smem + G4H_ * KR;             // [64][KR]  f16 A-operand rows
  float*    xs   = (float*)(smem + G4H_ * KR + B_ * KR);  // [64] (layer 0 only)

  const int tid  = threadIdx.x;
  const int wave = tid >> 5, lane = tid & 31;
  const int s  = wave >> 1, mh = wave & 1;
  const int lh = lane >> 4, ln = lane & 15;
  const int j  = s * 16 + ln;                    // owned hidden index

  // Per-thread async-copy geometry (layer 1): 32 bytes of h0_t per thread.
  const int gi = tid * 16;                       // halves within [64][128]
  const int cb = gi >> 7, ck = gi & 127;
  const unsigned lds_h0 =
      (unsigned)(size_t)(hbuf + cb * KR + 128 + ck);  // low 32 bits = LDS addr
  const unsigned voff = (unsigned)(tid * 32);         // byte offset in h0 frame

  // One-time: transpose weights f32->f16 into LDS (B tiles become contiguous).
  for (int idx = tid; idx < H_ * G4H_; idx += 512) {
    int k = idx >> 9, col = idx & 511;           // Wh[k][col]
    whT[col * KR + k] = (_Float16)Wh[idx];
  }
  if (LAYER == 1) {
    for (int idx = tid; idx < H_ * G4H_; idx += 512) {
      int k = idx >> 9, col = idx & 511;         // Wx1[k][col]
      whT[col * KR + 128 + k] = (_Float16)Wx[idx];
    }
  }
  // h(=0) region
  for (int idx = tid; idx < B_ * H_; idx += 512) {
    int b = idx >> 7, k = idx & 127;
    hbuf[b * KR + k] = (_Float16)0.0f;
  }

  float bias_r[4], wx0_r[4];
#pragma unroll
  for (int g = 0; g < 4; ++g) {
    bias_r[g] = bias[g * 128 + j];
    wx0_r[g]  = (LAYER == 0) ? Wx[g * 128 + j] : 0.0f;
  }
  F8 cst[2] = {};   // cell state, registers for whole sequence
  F8 hsum[2] = {};  // (layer 1) running sum for mean aggregation

  // Stage step-0 auxiliary data
  if (LAYER == 0) {
    if (tid < B_) xs[tid] = xnT[tid];
  } else {
    async_copy32(lds_h0, voff, (unsigned long long)(size_t)h0in);
    async_wait0();
  }
  __syncthreads();

  for (int t = 0; t < T_; ++t) {
    F8 acc[2][4] = {};
#pragma unroll
    for (int kt = 0; kt < KT; ++kt) {
      const int ko = kt * 32 + lh * 8;
      v16h a0 = load_a_tile(hbuf + (mh * 32 + 0 * 16 + ln) * KR + ko);
      v16h a1 = load_a_tile(hbuf + (mh * 32 + 1 * 16 + ln) * KR + ko);
#pragma unroll
      for (int g = 0; g < 4; ++g) {
        // B tile (32x16 f16): N = lane%16, K = kt*32 + lh*8 + 0..7, then kt*32 + 16 + lh*8 + 0..7
        v16h bt = load_a_tile(whT + (g * 128 + j) * KR + kt * 32 + lh * 8);
        acc[0][g].v = __builtin_amdgcn_wmma_f32_16x16x32_f16(
            false, a0, false, bt, (short)0, acc[0][g].v, false, false);
        acc[1][g].v = __builtin_amdgcn_wmma_f32_16x16x32_f16(
            false, a1, false, bt, (short)0, acc[1][g].v, false, false);
      }
    }

    F8 hreg[2];
#pragma unroll
    for (int m = 0; m < 2; ++m) {
#pragma unroll
      for (int e = 0; e < 8; ++e) {
        float xi = acc[m][0].f[e] + bias_r[0];
        float xf = acc[m][1].f[e] + bias_r[1];
        float xg = acc[m][2].f[e] + bias_r[2];
        float xo = acc[m][3].f[e] + bias_r[3];
        if (LAYER == 0) {
          int b = mh * 32 + m * 16 + lh * 8 + e;
          float xv = xs[b];
          xi += xv * wx0_r[0]; xf += xv * wx0_r[1];
          xg += xv * wx0_r[2]; xo += xv * wx0_r[3];
        }
        float cc = sigf(xf) * cst[m].f[e] + sigf(xi) * tanhf(xg);
        float hh = sigf(xo) * tanhf(cc);
        cst[m].f[e]  = cc;
        hreg[m].f[e] = hh;
        if (LAYER == 1) hsum[m].f[e] += hh;
      }
    }
    __syncthreads();  // all hbuf reads of step t complete

    // Kick off next step's h0_t DMA immediately (overlaps the h publish below).
    if (LAYER == 1 && t + 1 < T_) {
      const _Float16* gsrc = h0in + (size_t)(t + 1) * (B_ * H_);
      async_copy32(lds_h0, voff, (unsigned long long)(size_t)gsrc);
    }

    // publish h_t (cols 0..127) + stream to global (layer 0)
#pragma unroll
    for (int m = 0; m < 2; ++m) {
#pragma unroll
      for (int e = 0; e < 8; ++e) {
        int b = mh * 32 + m * 16 + lh * 8 + e;
        _Float16 hv = (_Float16)hreg[m].f[e];
        hbuf[b * KR + j] = hv;
        if (LAYER == 0) h0out[(size_t)t * (B_ * H_) + b * H_ + j] = hv;
      }
    }
    // stage step t+1 scalar input (layer 0); prefetch t+2 into L2/WGP$
    if (LAYER == 0 && t + 1 < T_) {
      if (tid < B_) {
        xs[tid] = xnT[(t + 1) * B_ + tid];
        if (t + 2 < T_) __builtin_prefetch(&xnT[(t + 2) * B_ + tid], 0, 3);
      }
    }
    if (LAYER == 1 && t + 1 < T_) async_wait0();  // DMA done before republish
    __syncthreads();
  }

  if (LAYER == 1) {
#pragma unroll
    for (int m = 0; m < 2; ++m) {
#pragma unroll
      for (int e = 0; e < 8; ++e) {
        int b = mh * 32 + m * 16 + lh * 8 + e;
        feat[b * H_ + j] = hsum[m].f[e] * (1.0f / T_);
      }
    }
  }
}

// ---------------------------------------------------------------------------
// Kernel 3: MLP head (tiny: ~17 MFLOP). Single workgroup, LDS ping-pong.
// out[0:64] = final prediction; out[64:64+64*192] = x_cache = [feat, ms_proj].
// ---------------------------------------------------------------------------
__global__ void head_kernel(const float* __restrict__ feat,
                            const float* __restrict__ mu,
                            const float* __restrict__ sd,
                            const float* __restrict__ theta,
                            const float* __restrict__ W_ms, const float* __restrict__ b_ms,
                            const float* __restrict__ W_th, const float* __restrict__ b_th,
                            const float* __restrict__ W_l1, const float* __restrict__ b_l1,
                            const float* __restrict__ W_l2, const float* __restrict__ b_l2,
                            const float* __restrict__ W_out, const float* __restrict__ b_out,
                            float* __restrict__ out) {
  extern __shared__ float zsm[];
  float* z0 = zsm;           // [64][256]
  float* z1 = zsm + 64 * 256;
  const int tid = threadIdx.x;

  // z = [feat(128), ms_proj(64), th(64)] ; also emit x_cache = z[:,0:192]
  for (int idx = tid; idx < 64 * 256; idx += 256) {
    int b = idx >> 8, c = idx & 255;
    float v;
    if (c < 128) {
      v = feat[b * H_ + c];
    } else if (c < 192) {
      int i = c - 128;
      v = mu[b] * W_ms[i] + sd[b] * W_ms[INC_ + i] + b_ms[i];
    } else {
      int i = c - 192;
      v = b_th[i];
#pragma unroll
      for (int d = 0; d < 4; ++d) v += theta[b * 4 + d] * W_th[d * INC_ + i];
    }
    z0[idx] = v;
    if (c < 192) out[64 + b * 192 + c] = v;  // x_cache
  }
  __syncthreads();

  for (int idx = tid; idx < 64 * L1_; idx += 256) {
    int b = idx >> 8, n = idx & 255;
    float a = b_l1[n];
#pragma unroll 8
    for (int k = 0; k < 256; ++k) a += z0[b * 256 + k] * W_l1[k * L1_ + n];
    z1[idx] = eluf(a);
  }
  __syncthreads();

  for (int idx = tid; idx < 64 * L2_; idx += 256) {
    int b = idx >> 8, n = idx & 255;
    float a = b_l2[n];
#pragma unroll 8
    for (int k = 0; k < 256; ++k) a += z1[b * 256 + k] * W_l2[k * L2_ + n];
    z0[idx] = eluf(a);
  }
  __syncthreads();

  if (tid < B_) {
    float a = b_out[0];
#pragma unroll 8
    for (int n = 0; n < 256; ++n) a += z0[tid * 256 + n] * W_out[n];
    out[tid] = a;
  }
}

// ---------------------------------------------------------------------------
extern "C" void kernel_launch(void* const* d_in, const int* in_sizes, int n_in,
                              void* d_out, int out_size, void* d_ws, size_t ws_size,
                              hipStream_t stream) {
  const float* x     = (const float*)d_in[0];
  const float* theta = (const float*)d_in[1];
  const float* Wx0   = (const float*)d_in[2];
  const float* Wh0   = (const float*)d_in[3];
  const float* b0    = (const float*)d_in[4];
  const float* Wx1   = (const float*)d_in[5];
  const float* Wh1   = (const float*)d_in[6];
  const float* b1    = (const float*)d_in[7];
  const float* W_ms  = (const float*)d_in[8];
  const float* b_ms  = (const float*)d_in[9];
  const float* W_th  = (const float*)d_in[10];
  const float* b_th  = (const float*)d_in[11];
  const float* W_l1  = (const float*)d_in[12];
  const float* b_l1  = (const float*)d_in[13];
  const float* W_l2  = (const float*)d_in[14];
  const float* b_l2  = (const float*)d_in[15];
  const float* W_out = (const float*)d_in[16];
  const float* b_out = (const float*)d_in[17];

  // Workspace layout (floats): mu[64] | sd[64] | rstd[64] | pad | feat[64*128]
  //                           | xnT[4096*64] | h0seq (f16, 4096*64*128)
  float* ws   = (float*)d_ws;
  float* mu   = ws;
  float* sd   = ws + 64;
  float* rstd = ws + 128;
  float* feat = ws + 256;
  float* xnT  = ws + 256 + B_ * H_;
  _Float16* h0seq = (_Float16*)(ws + 256 + B_ * H_ + T_ * B_);

  const int SH0 = (G4H_ * 128 + B_ * 128) * (int)sizeof(_Float16) + B_ * (int)sizeof(float);
  const int SH1 = (G4H_ * 256 + B_ * 256) * (int)sizeof(_Float16);
  const int SHH = 2 * 64 * 256 * (int)sizeof(float);

  hipFuncSetAttribute(reinterpret_cast<const void*>(lstm_kernel<0>),
                      hipFuncAttributeMaxDynamicSharedMemorySize, SH0);
  hipFuncSetAttribute(reinterpret_cast<const void*>(lstm_kernel<1>),
                      hipFuncAttributeMaxDynamicSharedMemorySize, SH1);
  hipFuncSetAttribute(reinterpret_cast<const void*>(head_kernel),
                      hipFuncAttributeMaxDynamicSharedMemorySize, SHH);

  stats_kernel<<<B_, 256, 0, stream>>>(x, mu, sd, rstd, xnT);
  lstm_kernel<0><<<1, 512, SH0, stream>>>(xnT, h0seq, Wx0, Wh0, b0, h0seq, feat);
  lstm_kernel<1><<<1, 512, SH1, stream>>>(xnT, h0seq, Wx1, Wh1, b1, h0seq, feat);
  head_kernel<<<1, 256, SHH, stream>>>(feat, mu, sd, theta, W_ms, b_ms, W_th, b_th,
                                       W_l1, b_l1, W_l2, b_l2, W_out, b_out,
                                       (float*)d_out);
}
